// S4Layer_79706003079906
// MI455X (gfx1250) — matrix operand order repaired and checked by reading, not compiled, not run on hardware
//
#include <hip/hip_runtime.h>
#include <hip/hip_bf16.h>

typedef _Float16 h16v __attribute__((ext_vector_type(16)));
typedef _Float16 h8v  __attribute__((ext_vector_type(8)));
typedef float    f8v  __attribute__((ext_vector_type(8)));

#define S4_B   2
#define S4_L   512
#define S4_DM  256
#define S4_DS  64
#define UPITCH 40   // halves; 80B rows: 16B-aligned b128 ops, 20-bank stride (conflict-free over 16 rows)

// ---------------------------------------------------------------------------
// Kernel 1: M = I + dt*A^T (row-vector update h' = h*M), Mp = M^16 via LDS squaring
// ---------------------------------------------------------------------------
__global__ __launch_bounds__(256) void s4_build_M(const float* __restrict__ A,
                                                  const float* __restrict__ log_dt,
                                                  float* __restrict__ Mout,
                                                  float* __restrict__ Mpout) {
  __shared__ float P[64 * 64];
  __shared__ float Q[64 * 64];
  const float dt = expf(log_dt[0]);
  const int t = threadIdx.x;
  for (int idx = t; idx < 4096; idx += 256) {
    const int j = idx >> 6, n = idx & 63;
    const float m = dt * A[n * 64 + j] + ((j == n) ? 1.0f : 0.0f);
    Mout[idx] = m;
    P[idx] = m;
  }
  __syncthreads();
  for (int it = 0; it < 4; ++it) {            // P <- P^2, four times => M^16
    for (int idx = t; idx < 4096; idx += 256) {
      const int j = idx >> 6, n = idx & 63;
      float acc = 0.0f;
      #pragma unroll 8
      for (int c = 0; c < 64; ++c) acc += P[j * 64 + c] * P[c * 64 + n];
      Q[idx] = acc;
    }
    __syncthreads();
    for (int idx = t; idx < 4096; idx += 256) P[idx] = Q[idx];
    __syncthreads();
  }
  for (int idx = t; idx < 4096; idx += 256) Mpout[idx] = P[idx];
}

// ---------------------------------------------------------------------------
// Kernel 2: Kd[k][m] = dt * (v_k . C[m,:]) in f16, E[m] = 2*Kd[0][m] + 2*D[m].
// 32 blocks; block j jumps to v_{16j} via M^16, then walks 16 steps with M.
// ---------------------------------------------------------------------------
__global__ __launch_bounds__(256) void s4_build_K(const float* __restrict__ Bv,
                                                  const float* __restrict__ C,
                                                  const float* __restrict__ Dv,
                                                  const float* __restrict__ log_dt,
                                                  const float* __restrict__ M,
                                                  const float* __restrict__ Mp,
                                                  _Float16* __restrict__ Kd,
                                                  _Float16* __restrict__ Eh) {
  __shared__ float v[64];
  __shared__ float vn[64];
  const int t = threadIdx.x;
  const float dt = expf(log_dt[0]);
  if (t < 64) v[t] = Bv[t];
  __syncthreads();
  for (int it = 0; it < (int)blockIdx.x; ++it) {   // v <- B^T * (M^16)^blockIdx.x
    if (t < 64) {
      float a = 0.0f;
      #pragma unroll 8
      for (int c = 0; c < 64; ++c) a += v[c] * Mp[c * 64 + t];
      vn[t] = a;
    }
    __syncthreads();
    if (t < 64) v[t] = vn[t];
    __syncthreads();
  }
  for (int tt = 0; tt < 16; ++tt) {
    const int k = blockIdx.x * 16 + tt;
    float a = 0.0f;                       // thread t == output channel m
    #pragma unroll 8
    for (int c = 0; c < 64; ++c) a += v[c] * C[t * 64 + c];
    const float kd = dt * a;
    Kd[k * 256 + t] = (_Float16)kd;
    if (k == 0) Eh[t] = (_Float16)(2.0f * kd + 2.0f * Dv[t]);
    if (t < 64) {                          // v <- v * M
      float a2 = 0.0f;
      #pragma unroll 8
      for (int c = 0; c < 64; ++c) a2 += v[c] * M[c * 64 + t];
      vn[t] = a2;
    }
    __syncthreads();
    if (t < 64) v[t] = vn[t];
    __syncthreads();
  }
}

// ---------------------------------------------------------------------------
// Kernel 3: per (b,l,dblk,mblk) a 128x128 x K=512 GEMM tile:
//   Y[d,m] = sum_s Uc[d,s] * W_l[s,m],  W_l row s = Kd[|l-s|] (E on diagonal)
// Software-pipelined: next chunk's global loads issue before the 8-WMMA chain,
// convert+ds_store after it, LDS ping-pong, one barrier per chunk.
// ---------------------------------------------------------------------------
__global__ __launch_bounds__(256) void s4_gemm(const float* __restrict__ x,
                                               const _Float16* __restrict__ Kd,
                                               const _Float16* __restrict__ Eh,
                                               float* __restrict__ out) {
  __shared__ __align__(16) _Float16 ldsU[2][128 * UPITCH];  // U^T chunk: [d][s]
  __shared__ __align__(16) _Float16 ldsW[2][128 * UPITCH];  // W_l chunk, transposed: [m][s]

  const int wg   = blockIdx.x;
  const int mblk = wg & 1;
  const int dblk = (wg >> 1) & 1;
  const int l    = (wg >> 2) & (S4_L - 1);
  const int b    = wg >> 11;
  const int t    = threadIdx.x;
  const int wave = t >> 5;
  const int lane = t & 31;
  const int half = lane >> 4;    // 0: lanes 0-15, 1: lanes 16-31
  const int l16  = lane & 15;
  const int mbase = mblk * 128;

  const int cidx = t & 127;         // staging column (d or m)
  const int sh   = (t >> 7) * 16;   // staging s-half: 16 consecutive s per thread

  const float*    xcol = x + (size_t)b * S4_L * S4_DM + dblk * 128 + cidx;  // + s*256
  const _Float16* krow = Kd + mbase + cidx;                                 // + kk*256

  f8v acc[8] = {};

  // ---- prologue: stage chunk 0 ----
  {
    float uf[16];
    _Float16 wf[16];
    #pragma unroll
    for (int i = 0; i < 16; ++i) uf[i] = xcol[(size_t)(sh + i) * S4_DM];
    #pragma unroll
    for (int i = 0; i < 16; ++i) {
      const int s = sh + i;
      int kk = l - s; if (kk < 0) kk = -kk;
      const _Float16* rp = (s == l) ? (Eh + mbase + cidx) : (krow + kk * 256);
      wf[i] = *rp;
    }
    h8v u0, u1, w0, w1;
    #pragma unroll
    for (int i = 0; i < 8; ++i) {
      u0[i] = (_Float16)uf[i];     u1[i] = (_Float16)uf[8 + i];
      w0[i] = wf[i];               w1[i] = wf[8 + i];
    }
    *(h8v*)(&ldsU[0][cidx * UPITCH + sh])     = u0;
    *(h8v*)(&ldsU[0][cidx * UPITCH + sh + 8]) = u1;
    *(h8v*)(&ldsW[0][cidx * UPITCH + sh])     = w0;
    *(h8v*)(&ldsW[0][cidx * UPITCH + sh + 8]) = w1;
  }

  for (int kc = 0; kc < 16; ++kc) {
    const int cur = kc & 1, nxt = cur ^ 1;
    // ---- issue next chunk's global loads (latency hidden under WMMA chain) ----
    float uf[16];
    _Float16 wf[16];
    if (kc < 15) {
      const int k0n = (kc + 1) * 32;
      #pragma unroll
      for (int i = 0; i < 16; ++i) uf[i] = xcol[(size_t)(k0n + sh + i) * S4_DM];
      #pragma unroll
      for (int i = 0; i < 16; ++i) {
        const int s = k0n + sh + i;
        int kk = l - s; if (kk < 0) kk = -kk;
        const _Float16* rp = (s == l) ? (Eh + mbase + cidx) : (krow + kk * 256);
        wf[i] = *rp;
      }
    }
    __syncthreads();

    // ---- A fragment (16x32 f16: lane half selects K 0-15 vs 16-31, matching B) ----
    const int arow = wave * 16 + l16;
    const int ko   = half * 16;
    const h8v a0 = *(const h8v*)(&ldsU[cur][arow * UPITCH + ko]);
    const h8v a1 = *(const h8v*)(&ldsU[cur][arow * UPITCH + ko + 8]);
    const h16v af = __builtin_shufflevector(a0, a1, 0, 1, 2, 3, 4, 5, 6, 7,
                                            8, 9, 10, 11, 12, 13, 14, 15);
    // ---- 8 m-tiles: B fragment (32x16 f16: lane = column, half selects K 0-15 vs 16-31) ----
    #pragma unroll
    for (int tm = 0; tm < 8; ++tm) {
      const int brow = tm * 16 + l16;
      const int bo   = half * 16;
      const h8v b0 = *(const h8v*)(&ldsW[cur][brow * UPITCH + bo]);
      const h8v b1 = *(const h8v*)(&ldsW[cur][brow * UPITCH + bo + 8]);
      const h16v bf = __builtin_shufflevector(b0, b1, 0, 1, 2, 3, 4, 5, 6, 7,
                                              8, 9, 10, 11, 12, 13, 14, 15);
      acc[tm] = __builtin_amdgcn_wmma_f32_16x16x32_f16(
          /*neg_a=*/false, af, /*neg_b=*/false, bf,
          /*c_mod=*/(short)0, acc[tm], /*reuse_a=*/false, /*reuse_b=*/false);
    }

    // ---- convert + store next chunk into the other buffer ----
    if (kc < 15) {
      h8v u0, u1, w0, w1;
      #pragma unroll
      for (int i = 0; i < 8; ++i) {
        u0[i] = (_Float16)uf[i];     u1[i] = (_Float16)uf[8 + i];
        w0[i] = wf[i];               w1[i] = wf[8 + i];
      }
      *(h8v*)(&ldsU[nxt][cidx * UPITCH + sh])     = u0;
      *(h8v*)(&ldsU[nxt][cidx * UPITCH + sh + 8]) = u1;
      *(h8v*)(&ldsW[nxt][cidx * UPITCH + sh])     = w0;
      *(h8v*)(&ldsW[nxt][cidx * UPITCH + sh + 8]) = w1;
    }
  }

  // ---- store: C/D layout VGPR r -> row r (lanes 0-15) / r+8 (lanes 16-31) ----
  // Non-temporal: output is write-once (268 MB), keep L2 for x/Kd reuse.
  const int d0 = dblk * 128 + wave * 16 + half * 8;
  const size_t base = ((size_t)b * S4_L + l) * S4_DM;
  #pragma unroll
  for (int tm = 0; tm < 8; ++tm) {
    const int m = mbase + tm * 16 + l16;
    #pragma unroll
    for (int r = 0; r < 8; ++r) {
      __builtin_nontemporal_store(acc[tm][r], &out[(base + (size_t)(d0 + r)) * S4_DM + m]);
    }
  }
}

// ---------------------------------------------------------------------------
extern "C" void kernel_launch(void* const* d_in, const int* in_sizes, int n_in,
                              void* d_out, int out_size, void* d_ws, size_t ws_size,
                              hipStream_t stream) {
  const float* x      = (const float*)d_in[0];
  const float* A      = (const float*)d_in[1];
  const float* Bv     = (const float*)d_in[2];
  const float* C      = (const float*)d_in[3];
  const float* Dv     = (const float*)d_in[4];
  const float* log_dt = (const float*)d_in[5];
  float* out = (float*)d_out;

  // workspace carve-up (all deterministic, rewritten every call)
  char* w = (char*)d_ws;
  _Float16* Kd = (_Float16*)(w);                 // 512*256 halves = 262144 B
  _Float16* Eh = (_Float16*)(w + 262144);        // 256 halves    = 512 B
  float*    M  = (float*)(w + 262656);           // 64*64 f32     = 16384 B
  float*    Mp = (float*)(w + 279040);           // 64*64 f32     = 16384 B

  s4_build_M<<<1, 256, 0, stream>>>(A, log_dt, M, Mp);
  s4_build_K<<<32, 256, 0, stream>>>(Bv, C, Dv, log_dt, M, Mp, Kd, Eh);
  s4_gemm<<<S4_B * S4_L * 2 * 2, 256, 0, stream>>>(x, Kd, Eh, out);
}
